// AdaptiveConv2_83373905150154
// MI455X (gfx1250) — matrix operand order repaired and checked by reading, not compiled, not run on hardware
//
#include <hip/hip_runtime.h>
#include <hip/hip_bf16.h>

typedef float     v8f  __attribute__((ext_vector_type(8)));
typedef _Float16  v8h  __attribute__((ext_vector_type(8)));
typedef _Float16  v16h __attribute__((ext_vector_type(16)));
typedef unsigned int v4u __attribute__((ext_vector_type(4)));
typedef int          v4i __attribute__((ext_vector_type(4)));
typedef int          v8i __attribute__((ext_vector_type(8)));

union H16 { v16h v; v8h h[2]; };

// Weight LDS row stride: 576 f16 data + TDM pad (4 DW per 32 DW) = 324 DW = 648 f16.
#define WROW    648
#define ISTRIDE 72    // 64 + 8 f16 pad per (row,col) channel vector

// ---------------------------------------------------------------------------
// NCHW f32 -> NHWC f16 one-time transpose of x (block per (n,y) row)
// ---------------------------------------------------------------------------
__global__ __launch_bounds__(256)
void nchw_to_nhwc(const float* __restrict__ xin, _Float16* __restrict__ xf)
{
    __shared__ _Float16 T[64 * 132];
    const int row = blockIdx.x;                // n*128 + y
    const int n = row >> 7, y = row & 127;
    for (int i = threadIdx.x; i < 64 * 128; i += 256) {
        int c = i >> 7, x = i & 127;
        T[c * 132 + x] = (_Float16)xin[(((size_t)n * 64 + c) * 128 + y) * 128 + x];
    }
    __syncthreads();
    for (int i = threadIdx.x; i < 128 * 8; i += 256) {
        int x = i >> 3, c8 = i & 7;
        v8h o;
        #pragma unroll
        for (int q = 0; q < 8; q++) o[q] = T[(c8 * 8 + q) * 132 + x];
        *((v8h*)(xf + (((size_t)row) * 128 + x) * 64 + c8 * 8)) = o;
    }
}

// ---------------------------------------------------------------------------
// Fold BN into weights/bias; f16 weights in tap-major K order.
// ---------------------------------------------------------------------------
__global__ __launch_bounds__(256)
void prep_weights(const float* __restrict__ w,  const float* __restrict__ b,
                  const float* __restrict__ g,  const float* __restrict__ be,
                  const float* __restrict__ mean_, const float* __restrict__ var_,
                  _Float16* __restrict__ wf, float* __restrict__ fbias, int cout)
{
    int o = blockIdx.x;
    if (o >= cout) return;
    float scale = g[o] * rsqrtf(var_[o] + 1e-5f);
    if (threadIdx.x == 0) fbias[o] = b[o] * scale + be[o] - mean_[o] * scale;
    for (int i = threadIdx.x; i < 576; i += 256) {
        int tap = i >> 6, cin = i & 63;
        wf[(size_t)o * 576 + i] = (_Float16)(w[((size_t)(o * 64 + cin)) * 9 + tap] * scale);
    }
}

// ---------------------------------------------------------------------------
// TDM: load [rows][576] f16 weight matrix from global into LDS, inserting
// 4 DWORDs of pad every 32 DWORDs (-> LDS row stride 648 f16, conflict-free).
// D# built per CDNA5 ISA §8.3-8.6 (count=1, type=2(image), data_size=2B,
// 2-D tile 576 x rows, tensor_dim0_stride=576, pad_interval=4, pad_amount=3).
// 6-arg builtin form (clang-23 / therock-10.0): extra int32x8 descriptor
// words are unused for a 2-D tile -> zero-filled.
// ---------------------------------------------------------------------------
__device__ inline void tdm_load_weights(const _Float16* wf, unsigned ldsOff, int rows)
{
    unsigned long long ga = (unsigned long long)(uintptr_t)wf;
    v4u g0;
    g0[0] = 1u;                                              // count=1, user mode
    g0[1] = ldsOff;                                          // lds_addr (bytes)
    g0[2] = (unsigned)(ga & 0xffffffffu);                    // global_addr[31:0]
    g0[3] = (unsigned)((ga >> 32) & 0x01ffffffu) | (2u << 30); // addr[56:32] | type=2
    v8i g1;
    g1[0] = (int)((1u << 16) | (1u << 20) | (4u << 22) | (3u << 25));
    //             data_size=2B  pad_en     pad_int=32DW  pad_amt=4DW ; wg_mask=0
    g1[1] = (int)(576u << 16);           // [15:0] atomic_bar=0, [31:16] tensor_dim0.lo=576
    g1[2] = (int)((unsigned)rows << 16); // [15:0] tensor_dim0.hi=0, [31:16] tensor_dim1.lo=rows
    g1[3] = (int)(576u << 16);           // [15:0] tensor_dim1.hi=0, [31:16] tile_dim0=576
    g1[4] = rows;                        // [15:0] tile_dim1=rows, [31:16] tile_dim2=0
    g1[5] = 576;                         // tensor_dim0_stride[31:0]
    g1[6] = 0;                           // stride0.hi, tensor_dim1_stride.lo (unused, 2D)
    g1[7] = 0;
    v4i g2 = {0, 0, 0, 0};
    v4i g3 = {0, 0, 0, 0};
    v8i g4 = {0, 0, 0, 0, 0, 0, 0, 0};
    __builtin_amdgcn_tensor_load_to_lds(g0, g1, g2, g3, g4, 0);
}

// ---------------------------------------------------------------------------
// Implicit-GEMM 3x3 conv + fused BN + tanh via v_wmma_f32_16x16x32_f16.
// Block: 64 out-ch x (4 rows x 64 px); full K=576 resident in LDS.
// Each wave: one 16-ch M-tile x 8 N-tiles (2 rows x 4 x-tiles) register-blocked
// -> A fragment loaded once per K-chunk, 8 WMMAs per 18 ds_load_b128.
// ---------------------------------------------------------------------------
__global__ __launch_bounds__(256)
void conv3x3_wmma(const _Float16* __restrict__ inNHWC,
                  const _Float16* __restrict__ wf,
                  const float*    __restrict__ fbias,
                  _Float16* __restrict__ outNHWC,
                  float*    __restrict__ outF32,
                  int mt, int storeF32)
{
    extern __shared__ _Float16 smem[];
    _Float16* Wlds = smem;                     // [mt*16][WROW] (TDM-padded)
    _Float16* Ilds = smem + 64 * WROW;         // [6 rows][66 cols][ISTRIDE]

    const int tid = threadIdx.x;
    const int x0 = blockIdx.x * 64;
    const int y0 = blockIdx.y * 4;
    const int nb = blockIdx.z;
    const int wave = tid >> 5;
    const int lane = tid & 31;

    // ---- weights via Tensor Data Mover (wave 0 issues; EXEC-independent) ----
    if (wave == 0) {
        tdm_load_weights(wf, (unsigned)(uintptr_t)(void*)Wlds, mt * 16);
        __builtin_amdgcn_s_wait_tensorcnt(0);
    }
    // ---- input tile: rows y0-1..y0+4, cols x0-1..x0+64, 64ch HWC ----
    for (int i = tid; i < 6 * 66 * 8; i += 256) {
        int c8 = i & 7, cr = i >> 3;
        int col = cr % 66, r = cr / 66;
        int gy = y0 - 1 + r, gx = x0 - 1 + col;
        v8h val;
        if ((unsigned)gy < 128u && (unsigned)gx < 128u) {
            val = *((const v8h*)(inNHWC + (((size_t)nb * 128 + gy) * 128 + gx) * 64 + c8 * 8));
        } else {
            #pragma unroll
            for (int q = 0; q < 8; q++) val[q] = (_Float16)0.f;
        }
        ((v8h*)(Ilds + ((size_t)(r * 66 + col)) * ISTRIDE))[c8] = val;
    }
    __syncthreads();

    const int hl   = lane >> 4;    // lane half (K-offset selector)
    const int mrow = lane & 15;    // A row within M-tile / pixel within N-tile

    const int groups = mt * 2;     // (M-tile) x (row-pair)
    for (int g = wave; g < groups; g += 8) {
        const int m0    = (g % mt) * 16;
        const int rbase = (g / mt) * 2;        // strip rows rbase, rbase+1

        v8f acc[8];
        #pragma unroll
        for (int t8 = 0; t8 < 8; t8++) acc[t8] = (v8f){0.f,0.f,0.f,0.f,0.f,0.f,0.f,0.f};

        const _Float16* wbase = Wlds + (size_t)(m0 + mrow) * WROW;

        #pragma unroll
        for (int ch = 0; ch < 18; ch++) {          // K = 576 = 18 chunks of 32
            const int tap = ch >> 1;
            const int dy = tap / 3, dx = tap % 3;
            const int cbase = (ch & 1) * 32 + 8 * hl;
            // A fragment (pad-aware offsets: +8 f16 per 64 f16)
            const int k1 = ch * 32 + 8 * hl;
            const int k2 = k1 + 16;
            H16 a;
            a.h[0] = *(const v8h*)(wbase + k1 + ((k1 >> 6) << 3));
            a.h[1] = *(const v8h*)(wbase + k2 + ((k2 >> 6) << 3));
            #pragma unroll
            for (int rr = 0; rr < 2; rr++) {
                const _Float16* ib = Ilds +
                    (size_t)((rbase + rr + dy) * 66 + mrow + dx) * ISTRIDE + cbase;
                #pragma unroll
                for (int j = 0; j < 4; j++) {
                    H16 b;
                    const v8h* ip = (const v8h*)(ib + j * 16 * ISTRIDE);
                    b.h[0] = ip[0];
                b.h[1] = ip[2];
                    acc[rr * 4 + j] = __builtin_amdgcn_wmma_f32_16x16x32_f16(
                        false, a.v, false, b.v, (short)0, acc[rr * 4 + j], false, false);
                }
            }
        }

        // ---- epilogue: +bias, tanh, store (lane owns 8 consecutive channels) ----
        const int chbase = m0 + 8 * hl;
        #pragma unroll
        for (int t8 = 0; t8 < 8; t8++) {
            const int gy = y0 + rbase + (t8 >> 2);
            const int gx = x0 + (t8 & 3) * 16 + mrow;
            if (!storeF32) {
                v8h o;
                #pragma unroll
                for (int r = 0; r < 8; r++)
                    o[r] = (_Float16)tanhf(acc[t8][r] + fbias[chbase + r]);
                *((v8h*)(outNHWC + (((size_t)nb * 128 + gy) * 128 + gx) * 64 + chbase)) = o;
            } else {
                float* op = outF32 + (((size_t)nb * 128 + gy) * 128 + gx) * 48 + chbase;
                #pragma unroll
                for (int r = 0; r < 8; r++)
                    op[r] = tanhf(acc[t8][r] + fbias[chbase + r]);
            }
        }
    }
}

// ---------------------------------------------------------------------------
// Per-pixel basis einsum + dynamic 3x3 conv. Output-store bound (201 MB).
// ---------------------------------------------------------------------------
__global__ __launch_bounds__(256)
void dynconv(const float* __restrict__ xin, const float* __restrict__ hlast,
             const float* __restrict__ bases, float* __restrict__ out)
{
    __shared__ float bco[64 * 54];             // [pixel][m*9+l]
    const int x0 = blockIdx.x * 64;
    const int y  = blockIdx.y;
    const int n  = blockIdx.z;
    const int tid = threadIdx.x;
    const int xl = tid & 63, grp = tid >> 6;

    if (grp == 0) {
        const float* hf = hlast + (((size_t)n * 128 + y) * 128 + (x0 + xl)) * 48;
        float bf[36];
        #pragma unroll
        for (int i = 0; i < 36; i++) bf[i] = hf[i];
        #pragma unroll
        for (int m = 0; m < 6; m++)
            #pragma unroll
            for (int l = 0; l < 9; l++) {
                float s = 0.f;
                #pragma unroll
                for (int k = 0; k < 6; k++) s += bf[m * 6 + k] * bases[k * 9 + l];
                bco[xl * 54 + m * 9 + l] = s;
            }
    }
    __syncthreads();

    const int x = x0 + xl;
    const float* bc = bco + xl * 54;
    for (int it = 0; it < 16; it++) {
        const int c = grp + it * 4;
        float px[9];
        #pragma unroll
        for (int dy = 0; dy < 3; dy++)
            #pragma unroll
            for (int dx = 0; dx < 3; dx++) {
                int gy = y + dy - 1, gx = x + dx - 1;
                float vv = 0.f;
                if ((unsigned)gy < 128u && (unsigned)gx < 128u)
                    vv = xin[(((size_t)n * 64 + c) * 128 + gy) * 128 + gx];
                px[dy * 3 + dx] = vv;
            }
        #pragma unroll
        for (int m = 0; m < 6; m++) {
            float s = 0.f;
            #pragma unroll
            for (int l = 0; l < 9; l++) s += bc[m * 9 + l] * px[l];
            out[(((size_t)n * 384 + (c * 6 + m)) * 128 + y) * 128 + x] = s;
        }
    }
}

// ---------------------------------------------------------------------------
extern "C" void kernel_launch(void* const* d_in, const int* in_sizes, int n_in,
                              void* d_out, int out_size, void* d_ws, size_t ws_size,
                              hipStream_t stream)
{
    (void)in_sizes; (void)n_in; (void)out_size; (void)ws_size;
    const float* x   = (const float*)d_in[0];
    const float* w0  = (const float*)d_in[1];
    const float* b0  = (const float*)d_in[2];
    const float* g0  = (const float*)d_in[3];
    const float* be0 = (const float*)d_in[4];
    const float* mu0 = (const float*)d_in[5];
    const float* vr0 = (const float*)d_in[6];
    const float* wm  = (const float*)d_in[7];
    const float* bm  = (const float*)d_in[8];
    const float* gm  = (const float*)d_in[9];
    const float* bem = (const float*)d_in[10];
    const float* mum = (const float*)d_in[11];
    const float* vrm = (const float*)d_in[12];
    const float* wl  = (const float*)d_in[13];
    const float* bl  = (const float*)d_in[14];
    const float* gl  = (const float*)d_in[15];
    const float* bel = (const float*)d_in[16];
    const float* mul = (const float*)d_in[17];
    const float* vrl = (const float*)d_in[18];
    const float* bas = (const float*)d_in[19];
    float* out = (float*)d_out;

    // ---- workspace carve (~76 MB) ----
    char* ws = (char*)d_ws;
    size_t off = 0;
    auto alloc = [&](size_t bytes) -> void* {
        void* p = (void*)(ws + off);
        off += (bytes + 255) & ~(size_t)255;
        return p;
    };
    _Float16* xf16  = (_Float16*)alloc((size_t)8 * 128 * 128 * 64 * 2);
    _Float16* actA  = (_Float16*)alloc((size_t)8 * 128 * 128 * 64 * 2);
    _Float16* actB  = (_Float16*)alloc((size_t)8 * 128 * 128 * 64 * 2);
    float*    hlast = (float*)   alloc((size_t)8 * 128 * 128 * 48 * 4);
    _Float16* wprep[7];
    float*    fb[7];
    for (int i = 0; i < 7; i++) {
        wprep[i] = (_Float16*)alloc((size_t)64 * 576 * 2);
        fb[i]    = (float*)   alloc(64 * 4);
    }

    // ---- weight prep (BN folded, tap-major f16) ----
    prep_weights<<<64, 256, 0, stream>>>(w0, b0, g0, be0, mu0, vr0, wprep[0], fb[0], 64);
    for (int i = 0; i < 5; i++)
        prep_weights<<<64, 256, 0, stream>>>(wm + (size_t)i * 64 * 64 * 9, bm + i * 64,
                                             gm + i * 64, bem + i * 64, mum + i * 64,
                                             vrm + i * 64, wprep[1 + i], fb[1 + i], 64);
    prep_weights<<<36, 256, 0, stream>>>(wl, bl, gl, bel, mul, vrl, wprep[6], fb[6], 36);

    // ---- x -> NHWC f16 ----
    nchw_to_nhwc<<<dim3(8 * 128), 256, 0, stream>>>(x, xf16);

    // ---- conv stack ----
    const size_t smem = (size_t)(64 * WROW + 6 * 66 * ISTRIDE) * sizeof(_Float16); // 139968 B
    dim3 cgrid(2, 32, 8);
    conv3x3_wmma<<<cgrid, 256, smem, stream>>>(xf16, wprep[0], fb[0], actA, nullptr, 4, 0);
    conv3x3_wmma<<<cgrid, 256, smem, stream>>>(actA, wprep[1], fb[1], actB, nullptr, 4, 0);
    conv3x3_wmma<<<cgrid, 256, smem, stream>>>(actB, wprep[2], fb[2], actA, nullptr, 4, 0);
    conv3x3_wmma<<<cgrid, 256, smem, stream>>>(actA, wprep[3], fb[3], actB, nullptr, 4, 0);
    conv3x3_wmma<<<cgrid, 256, smem, stream>>>(actB, wprep[4], fb[4], actA, nullptr, 4, 0);
    conv3x3_wmma<<<cgrid, 256, smem, stream>>>(actA, wprep[5], fb[5], actB, nullptr, 4, 0);
    conv3x3_wmma<<<cgrid, 256, smem, stream>>>(actB, wprep[6], fb[6], nullptr, hlast, 3, 1);

    // ---- dynamic conv (final output) ----
    dynconv<<<dim3(2, 128, 8), 256, 0, stream>>>(x, hlast, bas, out);
}
